// MultiHeadAttention_32521492365726
// MI455X (gfx1250) — matrix operand order repaired and checked by reading, not compiled, not run on hardware
//
#include <hip/hip_runtime.h>

// ---------------- problem constants ----------------
#define D_IN   1024
#define D_OUT  1024
#define NHEAD  16
#define HDIM   64
#define SEQ    2048
#define BATCH  2
#define MTOT   (BATCH * SEQ)      // 4096 rows

#define NEG_INF (-__builtin_inff())

typedef __attribute__((ext_vector_type(16))) __bf16 bf16x16;
typedef __attribute__((ext_vector_type(8)))  float  f32x8;
typedef __attribute__((ext_vector_type(4)))  int    v4i;

union Frag { bf16x16 v; unsigned u[8]; };

__device__ __forceinline__ unsigned short f2bf(float f) {
  unsigned u = __float_as_uint(f);
  return (unsigned short)((u + 0x7fffu + ((u >> 16) & 1u)) >> 16);  // RNE
}

__device__ __forceinline__ f32x8 wmma_bf16(const Frag& a, const Frag& b, f32x8 c) {
  return __builtin_amdgcn_wmma_f32_16x16x32_bf16(
      /*neg_a=*/false, a.v, /*neg_b=*/false, b.v,
      /*c_mod=*/(short)0, c, /*reuse_a=*/false, /*reuse_b=*/false);
}

// ------- CDNA5 async global->LDS (direct, no VGPR staging; ASYNCcnt-tracked) -------
#if __has_builtin(__builtin_amdgcn_global_load_async_to_lds_b128)
#define HAVE_ASYNC_LDS 1
#else
#define HAVE_ASYNC_LDS 0
#endif

__device__ __forceinline__ void async_b128(unsigned short* lds, const unsigned short* g) {
#if HAVE_ASYNC_LDS
  // prototype (from clang diagnostic): arg0 = v4i addrspace(1)*, then LDS ptr, offset, cpol
  __builtin_amdgcn_global_load_async_to_lds_b128(
      (__attribute__((address_space(1))) v4i*)g,
      (__attribute__((address_space(3))) v4i*)lds, /*offset=*/0, /*cpol=*/0);
#else
  *(uint4*)lds = *(const uint4*)g;   // fallback: VGPR-staged b128 copy
#endif
}

template <int N>
__device__ __forceinline__ void wait_asynccnt() {
#if __has_builtin(__builtin_amdgcn_s_wait_asynccnt)
  __builtin_amdgcn_s_wait_asynccnt(N);
#else
  asm volatile("s_wait_asynccnt %0" :: "i"(N) : "memory");
#endif
}

// ---------------- fp32 -> bf16 pack ----------------
__global__ __launch_bounds__(256) void cvt_f32_bf16(const float* __restrict__ in,
                                                    unsigned short* __restrict__ out,
                                                    int n4) {
  int i = blockIdx.x * 256 + threadIdx.x;
  if (i >= n4) return;
  float4 f = ((const float4*)in)[i];
  ushort4 o;
  o.x = f2bf(f.x); o.y = f2bf(f.y); o.z = f2bf(f.z); o.w = f2bf(f.w);
  ((ushort4*)out)[i] = o;
}

// ---------------- WMMA GEMM: Y[M,N] = X[M,K] * W[N,K]^T (+bias) ----------------
// Double-buffered: async-stage tile t+1 while WMMAs consume tile t.
#define GBM 128
#define GBN 64
#define GBK 32
#define GLD 40   // padded LDS K-stride: 80B = multiple of 16B for b128 async writes

template <bool BF16_OUT>
__global__ __launch_bounds__(256) void wmma_gemm(
    const unsigned short* __restrict__ X,   // bf16 bits [M,K]
    const unsigned short* __restrict__ W,   // bf16 bits [N,K]
    void* __restrict__ Yv,                  // bf16 [M,N] or f32 [M,N]
    const float* __restrict__ bias,         // f32 [N] (only if !BF16_OUT)
    int M, int N, int K) {
  __shared__ __attribute__((aligned(16))) unsigned short As[2][GBM * GLD];
  __shared__ __attribute__((aligned(16))) unsigned short Ws[2][GBN * GLD];
  const int tid  = threadIdx.x;
  const int lane = tid & 31, wave = tid >> 5;
  const int lm   = lane & 15, half = lane >> 4;
  const int m0   = blockIdx.x * GBM, n0 = blockIdx.y * GBN;
  f32x8 acc[4] = {};

  // stage one 128x32 A tile + 64x32 W tile: 3 b128 issues per thread (ASYNCcnt += 3/wave)
  auto stage = [&](int buf, int k0) {
    #pragma unroll
    for (int i = 0; i < 2; ++i) {          // A: 512 chunks of 8 bf16 (16B)
      int chunk = tid + (i << 8);          // 0..511
      int r = chunk >> 2;                  // 0..127
      int c = (chunk & 3) << 3;            // 0,8,16,24
      async_b128(&As[buf][r * GLD + c], &X[(size_t)(m0 + r) * K + k0 + c]);
    }
    {                                      // W: 256 chunks
      int r = tid >> 2;                    // 0..63
      int c = (tid & 3) << 3;
      async_b128(&Ws[buf][r * GLD + c], &W[(size_t)(n0 + r) * K + k0 + c]);
    }
  };

  const int NT = K / GBK;
  stage(0, 0);
  for (int kt = 0; kt < NT; ++kt) {
    const int cur = kt & 1;
    if (kt + 1 < NT) {
      stage(1 - cur, (kt + 1) * GBK);      // overlap next-tile fill with compute
      wait_asynccnt<3>();                  // oldest 3 (= current tile) complete, in order
    } else {
      wait_asynccnt<0>();
    }
    __syncthreads();
    // A fragment: ISA 16-bit A 16x32 layout. M=lm, K=(v>>2)*16 + half*8 + (v&3)*2
    Frag a;
    const int mrow = wave << 4;
    #pragma unroll
    for (int v = 0; v < 8; ++v) {
      int kk = ((v >> 2) << 4) + (half << 3) + ((v & 3) << 1);
      a.u[v] = *(const unsigned*)(&As[cur][(mrow + lm) * GLD + kk]);
    }
    #pragma unroll
    for (int nt = 0; nt < 4; ++nt) {
      // B fragment: N=lm, K=(v>>2)*16 + half*8 + (v&3)*2 + p ; B[k][n] = W[n][k] (rows K-contiguous)
      Frag b;
      #pragma unroll
      for (int v = 0; v < 8; ++v)
        b.u[v] = *(const unsigned*)(&Ws[cur][(nt * 16 + lm) * GLD + ((v >> 2) << 4) + (half << 3) + ((v & 3) << 1)]);
      acc[nt] = wmma_bf16(a, b, acc[nt]);
    }
    __syncthreads();                       // buf[cur] free for re-staging at kt+2
  }
  // C/D layout: M = v + 8*half, N = lm
  #pragma unroll
  for (int nt = 0; nt < 4; ++nt) {
    #pragma unroll
    for (int v = 0; v < 8; ++v) {
      const int row = m0 + (wave << 4) + v + (half << 3);
      const int col = n0 + nt * 16 + lm;
      if constexpr (BF16_OUT) {
        ((unsigned short*)Yv)[(size_t)row * N + col] = f2bf(acc[nt][v]);
      } else {
        ((float*)Yv)[(size_t)row * N + col] = acc[nt][v] + bias[col];
      }
    }
  }
}

// ---------------- causal flash attention (bf16 WMMA, f32 online softmax) ----------------
#define BQ    128
#define BKV   32
#define KS_LD 72   // 144B row stride: multiple of 16B for b128 async writes
#define VT_LD 36   // 32 + 4 pad
#define PS_LD 36

__global__ __launch_bounds__(256) void flash_attn(
    const unsigned short* __restrict__ Q,   // bf16 [B, SEQ, D_OUT]
    const unsigned short* __restrict__ K,
    const unsigned short* __restrict__ V,
    unsigned short* __restrict__ O) {
  __shared__ __attribute__((aligned(16))) unsigned short Ks[BKV * KS_LD];   // [kv][d]
  __shared__ __attribute__((aligned(16))) unsigned short Vt[HDIM * VT_LD];  // [d][kv]
  __shared__ __attribute__((aligned(16))) unsigned short Ps[8][16 * PS_LD]; // per-wave P
  const int qbi = blockIdx.x, h = blockIdx.y, b = blockIdx.z;
  const int tid  = threadIdx.x;
  const int lane = tid & 31, wave = tid >> 5;
  const int lm   = lane & 15, half = lane >> 4;
  const size_t base = ((size_t)b * SEQ) * D_OUT + (size_t)h * HDIM;
  const int qrow0 = qbi * BQ + (wave << 4);
  const float SCALE = 0.03125f;  // 1/sqrt(1024)

  // Q fragments live in registers for the whole K loop (2 k-steps cover hd=64)
  Frag qf[2];
  #pragma unroll
  for (int ks = 0; ks < 2; ++ks)
    #pragma unroll
    for (int v = 0; v < 8; ++v) {
      int kk = ks * 32 + ((v >> 2) << 4) + (half << 3) + ((v & 3) << 1);
      qf[ks].u[v] = *(const unsigned*)(&Q[base + (size_t)(qrow0 + lm) * D_OUT + kk]);
    }

  f32x8 accO[4] = {};
  float mrow[8], lrow[8];
  #pragma unroll
  for (int v = 0; v < 8; ++v) { mrow[v] = NEG_INF; lrow[v] = 0.f; }

  const int kbmax = (qbi * BQ + BQ - 1) / BKV;  // causal block limit (inclusive)
  for (int kb = 0; kb <= kbmax; ++kb) {
    const int kbase = kb * BKV;
    __syncthreads();
    {   // K tile: 32x64 bf16 = 256 b128 chunks, async direct-to-LDS, 1 per thread
      int r = tid >> 3;                 // kv row 0..31
      int c = (tid & 7) << 3;           // d 0,8,...,56
      async_b128(&Ks[r * KS_LD + c], &K[base + (size_t)(kbase + r) * D_OUT + c]);
    }
    #pragma unroll
    for (int i = 0; i < 2; ++i) {       // V tile: VGPR-staged transpose into Vt[d][kv]
      int chunk = tid + (i << 8);       // 0..511
      int r = chunk >> 4;               // kv row 0..31
      int c = (chunk & 15) << 2;        // d 0..60
      uint2 vv = *(const uint2*)(&V[base + (size_t)(kbase + r) * D_OUT + c]);
      Vt[(c + 0) * VT_LD + r] = (unsigned short)(vv.x & 0xffffu);
      Vt[(c + 1) * VT_LD + r] = (unsigned short)(vv.x >> 16);
      Vt[(c + 2) * VT_LD + r] = (unsigned short)(vv.y & 0xffffu);
      Vt[(c + 3) * VT_LD + r] = (unsigned short)(vv.y >> 16);
    }
    wait_asynccnt<0>();
    __syncthreads();

    if (kbase <= qrow0 + 15) {  // wave-uniform: skip tiles fully above the diagonal
      // S = Q * K^T  (16 x 32 per wave)
      f32x8 s[2] = {};
      #pragma unroll
      for (int ks = 0; ks < 2; ++ks) {
        #pragma unroll
        for (int nt = 0; nt < 2; ++nt) {
          Frag bfr;  // B[k=d][n=key] = Ks[key][d], pairs contiguous in d
          #pragma unroll
          for (int v = 0; v < 8; ++v)
            bfr.u[v] = *(const unsigned*)(&Ks[(nt * 16 + lm) * KS_LD + ks * 32 + ((v >> 2) << 4) + (half << 3) + ((v & 3) << 1)]);
          s[nt] = wmma_bf16(qf[ks], bfr, s[nt]);
        }
      }
      // online softmax, per row (row = qrow0 + v + 8*half; cols across 16-lane half)
      const bool needmask = (kbase + BKV - 1) > qrow0;
      #pragma unroll
      for (int v = 0; v < 8; ++v) {
        const int row = qrow0 + v + (half << 3);
        float x0 = s[0][v] * SCALE;
        float x1 = s[1][v] * SCALE;
        if (needmask) {
          if (kbase + lm > row)      x0 = NEG_INF;
          if (kbase + 16 + lm > row) x1 = NEG_INF;
        }
        float mx = fmaxf(x0, x1);
        #pragma unroll
        for (int off = 1; off < 16; off <<= 1)
          mx = fmaxf(mx, __shfl_xor(mx, off, 32));
        const float mnew = fmaxf(mrow[v], mx);
        const float corr = __expf(mrow[v] - mnew);
        const float p0 = __expf(x0 - mnew);
        const float p1 = __expf(x1 - mnew);
        float rs = p0 + p1;
        #pragma unroll
        for (int off = 1; off < 16; off <<= 1)
          rs += __shfl_xor(rs, off, 32);
        lrow[v] = lrow[v] * corr + rs;
        mrow[v] = mnew;
        #pragma unroll
        for (int nt = 0; nt < 4; ++nt) accO[nt][v] *= corr;
        const int prow = v + (half << 3);
        Ps[wave][prow * PS_LD + lm]      = f2bf(p0);
        Ps[wave][prow * PS_LD + 16 + lm] = f2bf(p1);
      }
      asm volatile("s_wait_dscnt 0" ::: "memory");  // in-wave LDS C->A relayout
      // P fragment (A-layout) and P * V
      Frag pf;
      #pragma unroll
      for (int v = 0; v < 8; ++v) {
        int kk = ((v >> 2) << 4) + (half << 3) + ((v & 3) << 1);
        pf.u[v] = *(const unsigned*)(&Ps[wave][lm * PS_LD + kk]);
      }
      #pragma unroll
      for (int nt = 0; nt < 4; ++nt) {
        Frag vf;  // B[k=key][n=d] from transposed Vt[d][key], pairs contiguous in key
        #pragma unroll
        for (int v = 0; v < 8; ++v)
          vf.u[v] = *(const unsigned*)(&Vt[(nt * 16 + lm) * VT_LD + ((v >> 2) << 4) + (half << 3) + ((v & 3) << 1)]);
        accO[nt] = wmma_bf16(pf, vf, accO[nt]);
      }
    }
  }
  // epilogue: ctx = accO / l  -> bf16
  #pragma unroll
  for (int nt = 0; nt < 4; ++nt) {
    #pragma unroll
    for (int v = 0; v < 8; ++v) {
      const int row = qrow0 + v + (half << 3);
      const int col = nt * 16 + lm;
      O[base + (size_t)row * D_OUT + col] = f2bf(accO[nt][v] / lrow[v]);
    }
  }
}

// ---------------- host orchestration ----------------
extern "C" void kernel_launch(void* const* d_in, const int* in_sizes, int n_in,
                              void* d_out, int out_size, void* d_ws, size_t ws_size,
                              hipStream_t stream) {
  const float* x   = (const float*)d_in[0];
  const float* w_k = (const float*)d_in[1];
  const float* w_q = (const float*)d_in[2];
  const float* w_v = (const float*)d_in[3];
  const float* w_o = (const float*)d_in[4];
  const float* b_o = (const float*)d_in[5];

  // workspace layout (bf16 bits), total 48 MB
  const size_t SZ_X = (size_t)MTOT * D_IN;     // 4M elems
  const size_t SZ_W = (size_t)D_OUT * D_IN;    // 1M elems
  unsigned short* xb  = (unsigned short*)d_ws;
  unsigned short* qb  = xb  + SZ_X;
  unsigned short* kb  = qb  + SZ_X;
  unsigned short* vb  = kb  + SZ_X;
  unsigned short* cb  = vb  + SZ_X;
  unsigned short* wkb = cb  + SZ_X;
  unsigned short* wqb = wkb + SZ_W;
  unsigned short* wvb = wqb + SZ_W;
  unsigned short* wob = wvb + SZ_W;

  // 1) pack to bf16
  cvt_f32_bf16<<<(int)(SZ_X / 4 / 256), 256, 0, stream>>>(x,   xb,  (int)(SZ_X / 4));
  cvt_f32_bf16<<<(int)(SZ_W / 4 / 256), 256, 0, stream>>>(w_k, wkb, (int)(SZ_W / 4));
  cvt_f32_bf16<<<(int)(SZ_W / 4 / 256), 256, 0, stream>>>(w_q, wqb, (int)(SZ_W / 4));
  cvt_f32_bf16<<<(int)(SZ_W / 4 / 256), 256, 0, stream>>>(w_v, wvb, (int)(SZ_W / 4));
  cvt_f32_bf16<<<(int)(SZ_W / 4 / 256), 256, 0, stream>>>(w_o, wob, (int)(SZ_W / 4));

  // 2) projections (reference swaps: q from w_k, k from w_q)
  dim3 ggrid(MTOT / GBM, D_OUT / GBN);
  wmma_gemm<true><<<ggrid, 256, 0, stream>>>(xb, wkb, qb, nullptr, MTOT, D_OUT, D_IN);
  wmma_gemm<true><<<ggrid, 256, 0, stream>>>(xb, wqb, kb, nullptr, MTOT, D_OUT, D_IN);
  wmma_gemm<true><<<ggrid, 256, 0, stream>>>(xb, wvb, vb, nullptr, MTOT, D_OUT, D_IN);

  // 3) causal flash attention per (q-block, head, batch)
  flash_attn<<<dim3(SEQ / BQ, NHEAD, BATCH), 256, 0, stream>>>(qb, kb, vb, cb);

  // 4) output projection + bias -> fp32 d_out
  wmma_gemm<false><<<ggrid, 256, 0, stream>>>(cb, wob, d_out, b_o, MTOT, D_OUT, D_OUT);
}
